// VectorQuantizerFixed_72756745994825
// MI455X (gfx1250) — matrix operand order repaired and checked by reading, not compiled, not run on hardware
//
#include <hip/hip_runtime.h>

// ---------------------------------------------------------------------------
// VQ-VAE vector quantizer, fused for MI455X (gfx1250, wave32, WMMA).
// inputs:  d_in[0] = inputs     [32,512,256] f32  (N=16384 rows, D=256)
//          d_in[1] = embeddings [8192,256]   f32  (K=8192 codes)
// out:     [quantized (4194304 f32)] [loss (1 f32)] [indices (16384 f32)]
// ---------------------------------------------------------------------------

typedef __attribute__((ext_vector_type(16))) _Float16 v16h;
typedef __attribute__((ext_vector_type(8)))  _Float16 v8h;
typedef __attribute__((ext_vector_type(8)))  float    v8f;
typedef __attribute__((ext_vector_type(4)))  float    v4f;

constexpr int EMB_D   = 256;
constexpr int N_CODES = 8192;
constexpr int N_ROWS  = 16384;      // 32*512
constexpr int LDS_STRIDE = 264;     // 256 + 8 halves pad -> conflict-free banks
constexpr int CHUNKS  = 4;          // codebook split across blockIdx.y
constexpr int TILES_PER_CHUNK = N_CODES / 16 / CHUNKS;   // 128
constexpr int CODES_PER_CHUNK = N_CODES / CHUNKS;        // 2048

// ---------------------------------------------------------------------------
// Kernel 1: codebook f32 -> f16, plus ||e||^2 per code. One wave per code row.
// ---------------------------------------------------------------------------
__global__ void __launch_bounds__(256)
vq_precompute(const float* __restrict__ emb,
              _Float16* __restrict__ emb_h,
              float* __restrict__ enorm) {
  const int wave = (blockIdx.x * blockDim.x + threadIdx.x) >> 5;
  const int lane = threadIdx.x & 31;
  if (wave >= N_CODES) return;
  const int col = lane * 8;
  const float* src = emb + (size_t)wave * EMB_D + col;
  v4f a = *(const v4f*)src;
  v4f b = *(const v4f*)(src + 4);
  v8h o;
  float s = 0.0f;
  #pragma unroll
  for (int i = 0; i < 4; ++i) { o[i]     = (_Float16)a[i]; s += a[i] * a[i]; }
  #pragma unroll
  for (int i = 0; i < 4; ++i) { o[4 + i] = (_Float16)b[i]; s += b[i] * b[i]; }
  *(v8h*)(emb_h + (size_t)wave * EMB_D + col) = o;
  #pragma unroll
  for (int m = 16; m >= 1; m >>= 1) s += __shfl_xor(s, m, 32);
  if (lane == 0) enorm[wave] = s;
}

// ---------------------------------------------------------------------------
// B-fragment load helper: two runs of 8 f16, 16 halves apart (two ds_load_b128) per lane.
// ---------------------------------------------------------------------------
__device__ __forceinline__ v16h ldfragB(const _Float16* brow, int j) {
  v8h lo = *(const v8h*)(brow + 32 * j);
  v8h hi = *(const v8h*)(brow + 32 * j + 16);
  v16h b;
  #pragma unroll
  for (int i = 0; i < 8; ++i) { b[i] = lo[i]; b[8 + i] = hi[i]; }
  return b;
}

// ---------------------------------------------------------------------------
// Kernel 2: fused distance GEMM + argmin over one codebook chunk.
// Block = 256 threads = 8 waves; each wave owns 16 X-rows (A resident in VGPRs).
// blockIdx.y selects a 2048-code chunk; per-(row,chunk) candidates go to ws.
// score(m,n) = ||e_n||^2 - 2 * dot(x_m, e_n)   (||x||^2 is row-constant)
// Double-buffered LDS filled by GLOBAL_LOAD_ASYNC_TO_LDS_B128 (ASYNCcnt),
// software-pipelined B fragments, dual WMMA accumulator chains.
// ---------------------------------------------------------------------------
__global__ void __launch_bounds__(256)
vq_argmin(const float* __restrict__ inp,
          const _Float16* __restrict__ emb_h,
          const float* __restrict__ enorm,
          float* __restrict__ cand_v,
          int* __restrict__ cand_i) {
  __shared__ __attribute__((aligned(16))) _Float16 sB[2][16 * LDS_STRIDE]; // 16.9 KB
  __shared__ __attribute__((aligned(16))) float sEn[CODES_PER_CHUNK];      //  8.0 KB
  const int tid  = threadIdx.x;
  const int lane = tid & 31;
  const int wave = tid >> 5;
  const int hlf  = lane >> 4;      // 0: lanes 0-15, 1: lanes 16-31
  const int l16  = lane & 15;
  const int row_base = (blockIdx.x * 8 + wave) * 16;
  const int chunk    = blockIdx.y;
  const int tile0    = chunk * TILES_PER_CHUNK;

  // Staging decomposition: 512 b128 transfers per tile, 2 per thread.
  const int st_row0 = tid >> 5,         st_col0 = (tid & 31) * 8;
  const int st_row1 = (tid + 256) >> 5, st_col1 = (tid & 31) * 8;
  const unsigned lds_off0[2] = {
      (unsigned)(size_t)&sB[0][st_row0 * LDS_STRIDE + st_col0],
      (unsigned)(size_t)&sB[1][st_row0 * LDS_STRIDE + st_col0]};
  const unsigned lds_off1[2] = {
      (unsigned)(size_t)&sB[0][st_row1 * LDS_STRIDE + st_col1],
      (unsigned)(size_t)&sB[1][st_row1 * LDS_STRIDE + st_col1]};

  // Stage this chunk's ||e||^2 into LDS once (2048 f32).
  #pragma unroll
  for (int i = tid; i < CODES_PER_CHUNK; i += 256)
    sEn[i] = enorm[chunk * CODES_PER_CHUNK + i];

  // --- Resident A fragments: 16 rows x 256 K of f16, converted on the fly.
  // 16-bit A 16x32 layout (ISA 7.12.2): lane half selects K-subblocks
  //   VGPRs 0-3: K = 8*hlf + 0..7 ; VGPRs 4-7: K = 16 + 8*hlf + 0..7
  const float* arow = inp + (size_t)(row_base + l16) * EMB_D;
  v16h afr[8];
  #pragma unroll
  for (int j = 0; j < 8; ++j) {
    const float* p0 = arow + 32 * j + 8 * hlf;
    const float* p1 = arow + 32 * j + 16 + 8 * hlf;
    v4f a0 = *(const v4f*)p0, a1 = *(const v4f*)(p0 + 4);
    v4f b0 = *(const v4f*)p1, b1 = *(const v4f*)(p1 + 4);
    #pragma unroll
    for (int i = 0; i < 4; ++i) {
      afr[j][i]      = (_Float16)a0[i];
      afr[j][4 + i]  = (_Float16)a1[i];
      afr[j][8 + i]  = (_Float16)b0[i];
      afr[j][12 + i] = (_Float16)b1[i];
    }
  }

  float minv[8];
  int   mini[8];
  #pragma unroll
  for (int r = 0; r < 8; ++r) { minv[r] = 3.402823466e38f; mini[r] = 0; }

  // Async stage of one 16-code x 256-f16 tile into LDS buffer `buf`.
  auto stage_async = [&](int buf, int tile) {
    const _Float16* gB = emb_h + (size_t)tile * 16 * EMB_D;
    const unsigned long long g0 =
        (unsigned long long)(gB + st_row0 * EMB_D + st_col0);
    const unsigned long long g1 =
        (unsigned long long)(gB + st_row1 * EMB_D + st_col1);
    asm volatile("global_load_async_to_lds_b128 %0, %1, off"
                 :: "v"(lds_off0[buf]), "v"(g0) : "memory");
    asm volatile("global_load_async_to_lds_b128 %0, %1, off"
                 :: "v"(lds_off1[buf]), "v"(g1) : "memory");
  };

  // Prologue: stage tile0 into buffer 0.
  stage_async(0, tile0);
  asm volatile("s_wait_asynccnt 0x0" ::: "memory");
  __syncthreads();

  for (int t = 0; t < TILES_PER_CHUNK; ++t) {
    const int tile = tile0 + t;
    const int cur  = t & 1, nxt = cur ^ 1;
    const bool have_next = (t + 1 < TILES_PER_CHUNK);

    // Kick off async fill of the other buffer; completes under the WMMA work.
    if (have_next) {
      stage_async(nxt, tile + 1);
      __builtin_prefetch(emb_h + (size_t)(tile + 2) * 16 * EMB_D + tid * 16,
                         0, 0);                       // global_prefetch_b8
    }

    // 8 WMMAs over K=256 (32 per step); B fragments software-pipelined so the
    // ds_load for step j+1 is in flight while WMMA j executes; two independent
    // accumulator chains. B 32x16 layout: lane = N col; elements 0-7 hold
    // K 8*hlf + 0..7, elements 8-15 hold K 16 + 8*hlf + 0..7 of each step.
    v8f acc0 = {}, acc1 = {};
    const _Float16* brow = &sB[cur][l16 * LDS_STRIDE + 8 * hlf];
    v16h bcur = ldfragB(brow, 0);
    #pragma unroll
    for (int j = 0; j < 8; ++j) {
      v16h bnxt = (j < 7) ? ldfragB(brow, j + 1) : bcur;
      if (j & 1)
        acc1 = __builtin_amdgcn_wmma_f32_16x16x32_f16(
            false, afr[j], false, bcur, (short)0, acc1, false, false);
      else
        acc0 = __builtin_amdgcn_wmma_f32_16x16x32_f16(
            false, afr[j], false, bcur, (short)0, acc0, false, false);
      bcur = bnxt;
    }

    // C layout: lane covers column n = l16; VGPR r covers row m = r + 8*hlf.
    const float en   = sEn[t * 16 + l16];
    const int   nidx = tile * 16 + l16;
    #pragma unroll
    for (int r = 0; r < 8; ++r) {
      const float s = en - 2.0f * (acc0[r] + acc1[r]);
      if (s < minv[r]) { minv[r] = s; mini[r] = nidx; }  // '<' keeps first min
    }

    // Publish the freshly staged buffer to all waves.
    if (have_next)
      asm volatile("s_wait_asynccnt 0x0" ::: "memory");
    __syncthreads();
  }

  // Final argmin across the 16 lanes of each half (xor masks stay in-half).
  #pragma unroll
  for (int r = 0; r < 8; ++r) {
    float v = minv[r];
    int   i = mini[r];
    #pragma unroll
    for (int m = 1; m <= 8; m <<= 1) {
      const float ov = __shfl_xor(v, m, 32);
      const int   oi = __shfl_xor(i, m, 32);
      if (ov < v || (ov == v && oi < i)) { v = ov; i = oi; }
    }
    if (l16 == 0) {
      const int row = row_base + hlf * 8 + r;
      cand_v[chunk * N_ROWS + row] = v;
      cand_i[chunk * N_ROWS + row] = i;
    }
  }
}

// ---------------------------------------------------------------------------
// Kernel 2b: combine per-chunk candidates (ascending chunk order keeps
// first-min semantics with strict '<').
// ---------------------------------------------------------------------------
__global__ void __launch_bounds__(256)
vq_argmin_reduce(const float* __restrict__ cand_v,
                 const int* __restrict__ cand_i,
                 int* __restrict__ out_idx,
                 float* __restrict__ out_idx_f) {
  const int row = blockIdx.x * 256 + threadIdx.x;
  float v = cand_v[row];
  int   i = cand_i[row];
  #pragma unroll
  for (int c = 1; c < CHUNKS; ++c) {
    const float ov = cand_v[c * N_ROWS + row];
    const int   oi = cand_i[c * N_ROWS + row];
    if (ov < v) { v = ov; i = oi; }
  }
  out_idx[row]   = i;
  out_idx_f[row] = (float)i;
}

// ---------------------------------------------------------------------------
// Kernel 3: gather quantized = emb_f32[idx], write output, per-row sum sq diff.
// ---------------------------------------------------------------------------
__global__ void __launch_bounds__(256)
vq_gather(const float* __restrict__ emb,
          const float* __restrict__ inp,
          const int* __restrict__ idx,
          float* __restrict__ out_q,
          float* __restrict__ partial) {
  __shared__ float red[256];
  const int row = blockIdx.x;
  const int d   = threadIdx.x;
  const int e   = idx[row];
  const float q = emb[(size_t)e * EMB_D + d];
  const float x = inp[(size_t)row * EMB_D + d];
  out_q[(size_t)row * EMB_D + d] = q;  // straight-through forward == quantized
  const float df = q - x;
  red[d] = df * df;
  __syncthreads();
  #pragma unroll
  for (int s = 128; s > 0; s >>= 1) {
    if (d < s) red[d] += red[d + s];
    __syncthreads();
  }
  if (d == 0) partial[row] = red[0];
}

// ---------------------------------------------------------------------------
// Kernel 4: deterministic final reduction. loss = 1.25 * mse
// (e_latent_loss == q_latent_loss numerically).
// ---------------------------------------------------------------------------
__global__ void __launch_bounds__(256)
vq_loss(const float* __restrict__ partial, float* __restrict__ out_loss) {
  __shared__ float red[256];
  const int t = threadIdx.x;
  float s = 0.0f;
  for (int i = t; i < N_ROWS; i += 256) s += partial[i];  // fixed order
  red[t] = s;
  __syncthreads();
  #pragma unroll
  for (int k = 128; k > 0; k >>= 1) {
    if (t < k) red[t] += red[t + k];
    __syncthreads();
  }
  if (t == 0) out_loss[0] = red[0] * 1.25f / (float)(N_ROWS * EMB_D);
}

// ---------------------------------------------------------------------------
extern "C" void kernel_launch(void* const* d_in, const int* in_sizes, int n_in,
                              void* d_out, int out_size, void* d_ws, size_t ws_size,
                              hipStream_t stream) {
  (void)in_sizes; (void)n_in; (void)out_size; (void)ws_size;
  const float* inp = (const float*)d_in[0];   // [16384, 256]
  const float* emb = (const float*)d_in[1];   // [8192, 256]
  float* out = (float*)d_out;

  // Workspace layout (~4.9 MB):
  char* ws = (char*)d_ws;
  _Float16* emb_h   = (_Float16*)ws;                          // 4,194,304 B
  float*    enorm   = (float*)(ws + 4194304);                 //    32,768 B
  int*      idx     = (int*)(ws + 4227072);                   //    65,536 B
  float*    partial = (float*)(ws + 4292608);                 //    65,536 B
  float*    cand_v  = (float*)(ws + 4358144);                 //   262,144 B
  int*      cand_i  = (int*)(ws + 4620288);                   //   262,144 B

  float* out_q     = out;                 // 4,194,304 f32
  float* out_loss  = out + 4194304;       // 1 f32
  float* out_idx_f = out + 4194305;       // 16,384 f32

  vq_precompute<<<N_CODES / 8, 256, 0, stream>>>(emb, emb_h, enorm);
  vq_argmin<<<dim3(N_ROWS / 128, CHUNKS), 256, 0, stream>>>(inp, emb_h, enorm,
                                                            cand_v, cand_i);
  vq_argmin_reduce<<<N_ROWS / 256, 256, 0, stream>>>(cand_v, cand_i, idx, out_idx_f);
  vq_gather<<<N_ROWS, 256, 0, stream>>>(emb, inp, idx, out_q, partial);
  vq_loss<<<1, 256, 0, stream>>>(partial, out_loss);
}
